// BiLSTM_88484916232709
// MI455X (gfx1250) — matrix operand order repaired and checked by reading, not compiled, not run on hardware
//
#include <hip/hip_runtime.h>

// ---------------------------------------------------------------------------
// Lattice-LSTM (NER BiLSTM with lexicon lattice) for MI455X / gfx1250.
//
// Strategy (latency-bound sequential scan):
//  * pack_B*  : parallel kernels packing f32 weights -> f16 WMMA B-fragment
//               order in scratch (zero-padded K), L2-resident thereafter.
//  * lattice_scan : ONE workgroup (16 wave32 waves) runs all 512 timesteps.
//               All matmuls are v_wmma_f32_16x16x32_f16. h/c history kept in
//               a 10-slot LDS ring (gaz_starts only reach 8 steps back).
//  * tag_argmax : parallel tag projection + argmax -> int32 tags.
// ---------------------------------------------------------------------------

typedef __attribute__((ext_vector_type(16))) _Float16 v16h;
typedef __attribute__((ext_vector_type(8)))  _Float16 v8h;
typedef __attribute__((ext_vector_type(8)))  float    v8f;

union Vfrag {
    v16h v;
    struct { v8h lo; v8h hi; } p;
};

// problem dims
constexpr int Bn   = 8;
constexpr int Tn   = 512;
constexpr int Kn   = 8;
constexpr int Hn   = 128;   // per-direction hidden
constexpr int H3   = 384;
constexpr int RING = 10;    // >= 9 needed (starts in [j-8, j-1])

// GEMM1: A1 [16 x 256]  (cols 0..99 x_j, 104..231 h_prev, rest 0)
//        B1 [512n x 256k] = rows [W_ih|W_hh] (0..383), [Wa_ih|0] (384..511)
constexpr int NT1 = 32, KT1 = 8, SA1 = 256, SO1 = 512;
// GEMM2: A2 [64 x 192]  (cols 0..49 ge, 64..191 h_s, rest 0)
//        B2 [384n x 192k] = [Ww_ih | Ww_hh]
constexpr int NT2 = 24, KT2 = 6, SA2 = 192, SO2 = 384;
// GEMM3: A3 = c_w [64 x 128], B3 = Wa_hh [128n x 128k]
constexpr int NT3 = 8,  KT3 = 4, SA3 = 128, SO3 = 128;

constexpr int B1_ELEMS = NT1 * KT1 * 32 * 16;  // 131072 halfs
constexpr int B2_ELEMS = NT2 * KT2 * 32 * 16;  //  73728 halfs
constexpr int B3_ELEMS = NT3 * KT3 * 32 * 16;  //  16384 halfs

__device__ __forceinline__ float sigm(float x) { return 1.0f / (1.0f + __expf(-x)); }

// ------------------------- weight packing kernels --------------------------
// Fragment element (tile t=(nt,kt), lane, e): n = 16*nt + (lane&15),
// k = 32*kt + 16*(lane>>4) + e.  Pads are written as 0 (keeps WMMA NaN-free).

__global__ void pack_B1(const float* __restrict__ W_ih, const float* __restrict__ W_hh,
                        const float* __restrict__ Wa_ih, _Float16* __restrict__ dst) {
    int idx = blockIdx.x * 256 + threadIdx.x;
    if (idx >= B1_ELEMS) return;
    int e = idx & 15, lane = (idx >> 4) & 31, t = idx >> 9;
    int nt = t >> 3, kt = t & 7;
    int n = 16 * nt + (lane & 15);
    int k = 32 * kt + 16 * (lane >> 4) + e;
    float v = 0.0f;
    if (n < H3) {
        if (k < 100)                    v = W_ih[n * 100 + k];
        else if (k >= 104 && k < 232)   v = W_hh[n * 128 + (k - 104)];
    } else {
        int na = n - H3;
        if (k < 100)                    v = Wa_ih[na * 100 + k];
    }
    dst[idx] = (_Float16)v;
}

__global__ void pack_B2(const float* __restrict__ Ww_ih, const float* __restrict__ Ww_hh,
                        _Float16* __restrict__ dst) {
    int idx = blockIdx.x * 256 + threadIdx.x;
    if (idx >= B2_ELEMS) return;
    int e = idx & 15, lane = (idx >> 4) & 31, t = idx >> 9;
    int nt = t / KT2, kt = t % KT2;
    int n = 16 * nt + (lane & 15);
    int k = 32 * kt + 16 * (lane >> 4) + e;
    float v = 0.0f;
    if (k < 50)                 v = Ww_ih[n * 50 + k];
    else if (k >= 64 && k < 192) v = Ww_hh[n * 128 + (k - 64)];
    dst[idx] = (_Float16)v;
}

__global__ void pack_B3(const float* __restrict__ Wa_hh, _Float16* __restrict__ dst) {
    int idx = blockIdx.x * 256 + threadIdx.x;
    if (idx >= B3_ELEMS) return;
    int e = idx & 15, lane = (idx >> 4) & 31, t = idx >> 9;
    int nt = t >> 2, kt = t & 3;
    int n = 16 * nt + (lane & 15);
    int k = 32 * kt + 16 * (lane >> 4) + e;
    dst[idx] = (_Float16)Wa_hh[n * 128 + k];
}

// ------------------------------ WMMA tile ----------------------------------
__device__ __forceinline__ void gemm_tile(const _Float16* __restrict__ Bp,
                                          const _Float16* Ap, _Float16* Op,
                                          int mt, int nt, int KT, int SA, int SO,
                                          int lane) {
    v8f acc = {};
    const int m  = 16 * mt + (lane & 15);
    const int kb = 16 * (lane >> 4);
    for (int kt = 0; kt < KT; ++kt) {
        Vfrag bf;
        bf.v = *(const v16h*)(Bp + ((size_t)(nt * KT + kt) * 32 + lane) * 16);
        if (kt + 1 < KT)
            __builtin_prefetch(Bp + ((size_t)(nt * KT + kt + 1) * 32 + lane) * 16);
        const _Float16* ap = Ap + m * SA + 32 * kt + kb;
        Vfrag af;
        af.p.lo = *(const v8h*)ap;
        af.p.hi = *(const v8h*)(ap + 8);
        acc = __builtin_amdgcn_wmma_f32_16x16x32_f16(false, af.v, false, bf.v,
                                                     (short)0, acc, false, false);
    }
    const int hi  = lane >> 4;
    const int col = 16 * nt + (lane & 15);
#pragma unroll
    for (int r = 0; r < 8; ++r)
        Op[(16 * mt + r + 8 * hi) * SO + col] = (_Float16)acc[r];
}

// ------------------------------ scan kernel --------------------------------
__global__ __launch_bounds__(512, 1)
void lattice_scan(const int* __restrict__ word_inputs, const int* __restrict__ biword_inputs,
                  const int* __restrict__ gaz_ids, const int* __restrict__ gaz_starts,
                  const int* __restrict__ gaz_mask,
                  const float* __restrict__ word_table, const float* __restrict__ biword_table,
                  const float* __restrict__ gaz_table,
                  const float* __restrict__ b_lstm, const float* __restrict__ b_alpha,
                  const float* __restrict__ b_word,
                  const _Float16* __restrict__ B1, const _Float16* __restrict__ B2,
                  const _Float16* __restrict__ B3,
                  float* __restrict__ fwd_out) {
    const int tid  = threadIdx.x;
    const int lane = tid & 31;
    const int wave = tid >> 5;

    __shared__ __align__(32) _Float16 sA1[16 * SA1];        //  8 KB
    __shared__ __align__(32) _Float16 sA2[64 * SA2];        // 24 KB
    __shared__ __align__(32) _Float16 sCw[64 * 128];        // 16 KB (A3 input)
    __shared__ __align__(32) _Float16 sOut1[16 * SO1];      // 16 KB (g3 | xa)
    __shared__ __align__(32) _Float16 sWg[64 * SO2];        // 48 KB (wg; reused for alpha)
    __shared__ __align__(32) _Float16 sHring[RING][Bn * Hn];// 20 KB
    __shared__ __align__(32) _Float16 sCring[RING][Bn * Hn];// 20 KB
    __shared__ int sS[64];
    __shared__ int sM[64];

    _Float16* sAlpha = sWg;  // alias: wg is dead once c_w is extracted

    // init: zero rings + static zero pads of A staging
    for (int i = tid; i < RING * Bn * Hn; i += 512) {
        sHring[0][i] = (_Float16)0.0f;
        sCring[0][i] = (_Float16)0.0f;
    }
    for (int i = tid; i < 16 * SA1; i += 512) sA1[i] = (_Float16)0.0f;
    for (int i = tid; i < 64 * SA2; i += 512) sA2[i] = (_Float16)0.0f;
    __syncthreads();

    for (int j = 0; j < Tn; ++j) {
        const int pslot = (j + RING - 1) % RING;  // slot of h/c at step j-1 (zeros at j=0)

        // ---- phase 0: stage A matrices (gathers) ----
        for (int idx = tid; idx < Bn * 100; idx += 512) {       // x_j rows
            int b = idx / 100, c = idx % 100;
            float v = (c < 50)
                ? word_table[(size_t)word_inputs[b * Tn + j] * 50 + c]
                : biword_table[(size_t)biword_inputs[b * Tn + j] * 50 + (c - 50)];
            sA1[b * SA1 + c] = (_Float16)v;
        }
        for (int idx = tid; idx < Bn * Hn; idx += 512) {        // h_prev
            int b = idx >> 7, c = idx & 127;
            sA1[b * SA1 + 104 + c] = sHring[pslot][b * Hn + c];
        }
        for (int idx = tid; idx < 64 * 50; idx += 512) {        // gaz embeddings
            int r = idx / 50, c = idx % 50;
            int b = r >> 3, k = r & 7;
            int gid = gaz_ids[(b * Tn + j) * Kn + k];
            sA2[r * SA2 + c] = (_Float16)gaz_table[(size_t)gid * 50 + c];
        }
        for (int idx = tid; idx < 64 * Hn; idx += 512) {        // h_s gather from ring
            int r = idx >> 7, c = idx & 127;
            int b = r >> 3, k = r & 7;
            int s = gaz_starts[(b * Tn + j) * Kn + k];
            sA2[r * SA2 + 64 + c] = sHring[s % RING][b * Hn + c];
        }
        if (tid < 64) {
            int b = tid >> 3, k = tid & 7;
            sS[tid] = gaz_starts[(b * Tn + j) * Kn + k];
            sM[tid] = gaz_mask[(b * Tn + j) * Kn + k];
        }
        __syncthreads();

        // ---- phase 1: WMMA GEMMs G1 (x/h -> g3|xa) and G2 (ge/h_s -> wg) ----
        for (int job = wave; job < NT1 + 4 * NT2; job += 16) {
            if (job < NT1) {
                gemm_tile(B1, sA1, sOut1, 0, job, KT1, SA1, SO1, lane);
            } else {
                int q = job - NT1;
                gemm_tile(B2, sA2, sWg, q / NT2, q % NT2, KT2, SA2, SO2, lane);
            }
        }
        __syncthreads();

        // ---- phase 2: word-cell elementwise: c_w = sig(f)*c_s + sig(i)*tanh(g) ----
        for (int idx = tid; idx < 64 * Hn; idx += 512) {
            int r = idx >> 7, h = idx & 127;
            float iw = (float)sWg[r * SO2 + h]       + b_word[h];
            float fw = (float)sWg[r * SO2 + 128 + h] + b_word[128 + h];
            float gw = (float)sWg[r * SO2 + 256 + h] + b_word[256 + h];
            int b = r >> 3;
            float cs = (float)sCring[sS[r] % RING][b * Hn + h];
            sCw[idx] = (_Float16)(sigm(fw) * cs + sigm(iw) * tanhf(gw));
        }
        __syncthreads();

        // ---- phase 3: WMMA GEMM G3 (c_w @ Wa_hh^T) ----
        for (int job = wave; job < 4 * NT3; job += 16)
            gemm_tile(B3, sCw, sAlpha, job >> 3, job & 7, KT3, SA3, SO3, lane);
        __syncthreads();

        // ---- phase 4: char gates + gated aggregation + state update ----
        for (int idx = tid; idx < Bn * Hn; idx += 512) {
            int b = idx >> 7, h = idx & 127;
            float iv = sigm((float)sOut1[b * SO1 + h]       + b_lstm[h]);
            float ov = sigm((float)sOut1[b * SO1 + 128 + h] + b_lstm[128 + h]);
            float gv = tanhf((float)sOut1[b * SO1 + 256 + h] + b_lstm[256 + h]);
            float xa = (float)sOut1[b * SO1 + 384 + h] + b_alpha[h];
            float sum_w = 0.0f, sum_wc = 0.0f;
            int any = 0;
#pragma unroll
            for (int k = 0; k < Kn; ++k) {
                int r = b * Kn + k;
                int mk = sM[r];
                float al = sigm((float)sAlpha[r * SO3 + h] + xa);
                float ww = mk ? __expf(al) : 0.0f;
                sum_w  += ww;
                sum_wc += ww * (float)sCw[r * 128 + h];
                any |= mk;
            }
            float wchar = __expf(iv);
            float cprev = (float)sCring[pslot][b * Hn + h];
            float csoft = (wchar * gv + sum_wc) / (wchar + sum_w);
            float ccpl  = (1.0f - iv) * cprev + iv * gv;
            float cnew  = any ? csoft : ccpl;
            float hnew  = ov * tanhf(cnew);
            int slot = j % RING;
            sCring[slot][b * Hn + h] = (_Float16)cnew;
            sHring[slot][b * Hn + h] = (_Float16)hnew;
            fwd_out[((size_t)b * Tn + j) * Hn + h] = hnew;
        }
        __syncthreads();
    }
}

// -------------------------- tag projection + argmax ------------------------
// lstm_out = [fwd, fwd]  =>  outs = fwd @ (W_tag[:, :128] + W_tag[:, 128:])^T + b_tag
__global__ void tag_argmax(const float* __restrict__ fwd, const float* __restrict__ W_tag,
                           const float* __restrict__ b_tag, const int* __restrict__ mask,
                           int* __restrict__ out) {
    int idx = blockIdx.x * blockDim.x + threadIdx.x;
    if (idx >= Bn * Tn) return;
    const float* h = fwd + (size_t)idx * Hn;
    float best = -3.4e38f;
    int   bi   = 0;
    for (int l = 0; l < 20; ++l) {
        float acc = b_tag[l];
        for (int c = 0; c < Hn; ++c)
            acc += h[c] * (W_tag[l * 256 + c] + W_tag[l * 256 + 128 + c]);
        if (acc > best) { best = acc; bi = l; }  // strict > keeps first max (jnp.argmax)
    }
    out[idx] = mask[idx] * bi;
}

// ------------------------------- launcher ----------------------------------
extern "C" void kernel_launch(void* const* d_in, const int* in_sizes, int n_in,
                              void* d_out, int out_size, void* d_ws, size_t ws_size,
                              hipStream_t stream) {
    const int*   word_inputs   = (const int*)d_in[0];
    const int*   biword_inputs = (const int*)d_in[1];
    const int*   gaz_ids       = (const int*)d_in[2];
    const int*   gaz_starts    = (const int*)d_in[3];
    const int*   gaz_mask      = (const int*)d_in[4];
    const int*   mask          = (const int*)d_in[5];
    const float* word_table    = (const float*)d_in[6];
    const float* biword_table  = (const float*)d_in[7];
    const float* gaz_table     = (const float*)d_in[8];
    const float* W_ih          = (const float*)d_in[9];
    const float* W_hh          = (const float*)d_in[10];
    const float* b_lstm        = (const float*)d_in[11];
    const float* Wa_ih         = (const float*)d_in[12];
    const float* Wa_hh         = (const float*)d_in[13];
    const float* b_alpha       = (const float*)d_in[14];
    const float* Ww_ih         = (const float*)d_in[15];
    const float* Ww_hh         = (const float*)d_in[16];
    const float* b_word        = (const float*)d_in[17];
    const float* W_tag         = (const float*)d_in[18];
    const float* b_tag         = (const float*)d_in[19];

    // scratch layout (all offsets 32B aligned); total ~2.45 MB
    char* ws = (char*)d_ws;
    _Float16* B1p = (_Float16*)(ws);
    _Float16* B2p = (_Float16*)(ws + (size_t)B1_ELEMS * 2);
    _Float16* B3p = (_Float16*)(ws + (size_t)(B1_ELEMS + B2_ELEMS) * 2);
    float*    fwd = (float*)   (ws + (size_t)(B1_ELEMS + B2_ELEMS + B3_ELEMS) * 2);

    pack_B1<<<(B1_ELEMS + 255) / 256, 256, 0, stream>>>(W_ih, W_hh, Wa_ih, B1p);
    pack_B2<<<(B2_ELEMS + 255) / 256, 256, 0, stream>>>(Ww_ih, Ww_hh, B2p);
    pack_B3<<<(B3_ELEMS + 255) / 256, 256, 0, stream>>>(Wa_hh, B3p);

    lattice_scan<<<1, 512, 0, stream>>>(word_inputs, biword_inputs, gaz_ids, gaz_starts,
                                        gaz_mask, word_table, biword_table, gaz_table,
                                        b_lstm, b_alpha, b_word, B1p, B2p, B3p, fwd);

    tag_argmax<<<(Bn * Tn + 255) / 256, 256, 0, stream>>>(fwd, W_tag, b_tag, mask,
                                                          (int*)d_out);
}
